// NonLocalBlock3D_40226663694537
// MI455X (gfx1250) — matrix operand order repaired and checked by reading, not compiled, not run on hardware
//
#include <hip/hip_runtime.h>
#include <hip/hip_bf16.h>

#define B_   8
#define C_   512
#define CI_  256
#define T_   8
#define H_   32
#define W_   32
#define N_   (T_*H_*W_)             // 8192
#define M_   ((T_/2)*(H_/2)*(W_/2)) // 1024
#define O3_  (3*CI_)                // 768
#define NTV_ (M_/32)                // 32 V tiles in attention pass 2

typedef _Float16 half_t;
typedef __attribute__((ext_vector_type(8)))  _Float16     v8h;
typedef __attribute__((ext_vector_type(16))) _Float16     v16h;
typedef __attribute__((ext_vector_type(8)))  float        v8f;
typedef __attribute__((ext_vector_type(4)))  unsigned int v4u;
typedef __attribute__((ext_vector_type(4)))  int          v4i;
typedef __attribute__((ext_vector_type(8)))  int          v8i;

#if defined(__has_builtin)
#if __has_builtin(__builtin_amdgcn_tensor_load_to_lds) && __has_builtin(__builtin_amdgcn_s_wait_tensorcnt)
#define HAVE_TDM 1
#endif
#endif
#ifndef HAVE_TDM
#define HAVE_TDM 0
#endif

union AFrag { v16h v; v8h h[2]; };

// A-matrix 16x32 f16 fragment (row-major source, K-contiguous rows).
// lane<16: K = k0..k0+15; lane>=16: K = k0+16..k0+31 (hh = lane>>4).
__device__ inline v16h load_a_frag(const half_t* base, int ldk, int row, int k0, int hh) {
  AFrag a;
  const half_t* p = base + (size_t)row * ldk + k0 + hh * 16;
  a.h[0] = *(const v8h*)(p);
  a.h[1] = *(const v8h*)(p + 8);
  return a.v;
}

// B-matrix 32x16 f16 fragment: lane<16 holds col, K=k0..k0+15; lane>=16 K=k0+16..k0+31.
__device__ inline v16h load_b_frag(const half_t* base, int ldk, int col, int k0, int hh) {
  return *(const v16h*)(base + (size_t)col * ldk + k0 + hh * 16);
}

__device__ inline v8f wmma_f32(v16h a, v16h b, v8f c) {
  return __builtin_amdgcn_wmma_f32_16x16x32_f16(false, a, false, b, (short)0, c, false, false);
}

#if HAVE_TDM
// Issue a 2D TDM load: tile_h rows x tile_w (2-byte) elements, global row stride
// row_stride_elts, into contiguous LDS rows of tile_w elements at lds_off.
__device__ inline void tdm_load_2d(unsigned int lds_off, const void* gptr,
                                   unsigned int tile_w, unsigned int tile_h,
                                   unsigned int row_stride_elts) {
  unsigned long long ga = (unsigned long long)(uintptr_t)gptr;
  v4u g0;
  g0[0] = 1u;                                    // count=1 (valid), user descriptor
  g0[1] = lds_off;                               // lds_addr (bytes)
  g0[2] = (unsigned int)ga;                      // global_addr[31:0]
  g0[3] = (unsigned int)((ga >> 32) & 0x01FFFFFFu) | (2u << 30);  // addr[56:32] | type=2
  v8i g1;
  g1[0] = (int)(1u << 16);                       // workgroup_mask=0, data_size=1 (2B)
  g1[1] = (int)(tile_w << 16);                   // tensor_dim0[15:0] @ bits[63:48]
  g1[2] = (int)((tile_w >> 16) | (tile_h << 16));// tensor_dim0 hi | tensor_dim1 lo
  g1[3] = (int)((tile_h >> 16) | (tile_w << 16));// tensor_dim1 hi | tile_dim0
  g1[4] = (int)(tile_h & 0xFFFFu);               // tile_dim1 | tile_dim2=0
  g1[5] = (int)row_stride_elts;                  // tensor_dim0_stride[31:0]
  g1[6] = 0;                                     // stride hi | tensor_dim1_stride lo
  g1[7] = 0;
  v4i z4 = {0, 0, 0, 0};
#if defined(__clang_major__) && __clang_major__ >= 23
  v8i z8 = {0, 0, 0, 0, 0, 0, 0, 0};
  __builtin_amdgcn_tensor_load_to_lds(g0, g1, z4, z4, z8, 0);
#else
  __builtin_amdgcn_tensor_load_to_lds(g0, g1, z4, z4, 0);
#endif
}
#endif

// ---------------------------------------------------------------- prep weights
__global__ void prep_kernel(const float* __restrict__ tw, const float* __restrict__ tb,
                            const float* __restrict__ pw, const float* __restrict__ pb,
                            const float* __restrict__ gw, const float* __restrict__ gb,
                            const float* __restrict__ wzw,
                            half_t* __restrict__ Wcat, float* __restrict__ biasCat,
                            half_t* __restrict__ wzh) {
  int idx = blockIdx.x * blockDim.x + threadIdx.x;
  if (idx < O3_ * C_) {
    int o = idx / C_, c = idx % C_;
    float v = (o < CI_) ? tw[o * C_ + c]
            : (o < 2 * CI_) ? pw[(o - CI_) * C_ + c]
                            : gw[(o - 2 * CI_) * C_ + c];
    Wcat[idx] = (half_t)v;
  }
  if (idx < C_ * CI_) wzh[idx] = (half_t)wzw[idx];
  if (idx < O3_) {
    biasCat[idx] = (idx < CI_) ? tb[idx] : (idx < 2 * CI_) ? pb[idx - CI_] : gb[idx - 2 * CI_];
  }
}

// ------------------------------------------------- x (b,C,N) f32 -> (b,N,C) f16
__global__ void transpose_kernel(const float* __restrict__ x, half_t* __restrict__ xh) {
  __shared__ float tile[32][33];
  int b = blockIdx.z;
  int n0 = blockIdx.x * 32, c0 = blockIdx.y * 32;
  int tx = threadIdx.x, ty = threadIdx.y;   // 32 x 8
#pragma unroll
  for (int j = 0; j < 4; ++j) {
    int c = ty + j * 8;
    tile[c][tx] = x[((size_t)b * C_ + c0 + c) * N_ + n0 + tx];
  }
  __syncthreads();
#pragma unroll
  for (int j = 0; j < 4; ++j) {
    int nr = ty + j * 8;
    xh[((size_t)b * N_ + n0 + nr) * C_ + c0 + tx] = (half_t)tile[tx][nr];
  }
}

// --------------------------------------- GEMM1: [768,512] x [512,N] per batch
// 64x64 wave tile; output rows depend only on blockIdx.x -> scalar segment branch.
// theta rows -> thetaT (b,N,CI); phi rows -> phiF (b,CI,N); g rows -> gF (b,CI,N)
__global__ __launch_bounds__(256, 1) void gemm_proj_kernel(
    const half_t* __restrict__ Wcat, const float* __restrict__ biasCat,
    const half_t* __restrict__ xh,
    half_t* __restrict__ thetaT, half_t* __restrict__ phiF, half_t* __restrict__ gF) {
  int lane = threadIdx.x & 31, wave = threadIdx.x >> 5;
  int r = lane & 15, hh = lane >> 4;
  int oB = blockIdx.x * 64;                 // scalar (uniform across block)
  int nB = blockIdx.y * 512 + wave * 64;    // waves partition n
  int b = blockIdx.z;
  const half_t* X = xh + (size_t)b * N_ * C_;   // (N, C) row-major, K contiguous

  v8f acc[4][4] = {};
  for (int kk = 0; kk < C_ / 32; ++kk) {
    int k0 = kk * 32;
    v16h a[4], bb[4];
#pragma unroll
    for (int to = 0; to < 4; ++to)
      a[to] = load_a_frag(Wcat, C_, oB + to * 16 + r, k0, hh);
#pragma unroll
    for (int tn = 0; tn < 4; ++tn)
      bb[tn] = load_b_frag(X, C_, nB + tn * 16 + r, k0, hh);
#pragma unroll
    for (int to = 0; to < 4; ++to)
#pragma unroll
      for (int tn = 0; tn < 4; ++tn)
        acc[to][tn] = wmma_f32(a[to], bb[tn], acc[to][tn]);
  }
  size_t tBase = (size_t)b * N_ * CI_;
  size_t fBase = (size_t)b * CI_ * N_;
#pragma unroll
  for (int to = 0; to < 4; ++to) {
    int o0 = oB + to * 16;                  // scalar -> scalar branch below
    if (o0 < CI_) {
#pragma unroll
      for (int tn = 0; tn < 4; ++tn)
#pragma unroll
        for (int i = 0; i < 8; ++i) {
          int o = o0 + i + hh * 8;
          int n = nB + tn * 16 + r;
          thetaT[tBase + (size_t)n * CI_ + o] = (half_t)(acc[to][tn][i] + biasCat[o]);
        }
    } else if (o0 < 2 * CI_) {
#pragma unroll
      for (int tn = 0; tn < 4; ++tn)
#pragma unroll
        for (int i = 0; i < 8; ++i) {
          int o = o0 + i + hh * 8;
          int n = nB + tn * 16 + r;
          phiF[fBase + (size_t)(o - CI_) * N_ + n] = (half_t)(acc[to][tn][i] + biasCat[o]);
        }
    } else {
#pragma unroll
      for (int tn = 0; tn < 4; ++tn)
#pragma unroll
        for (int i = 0; i < 8; ++i) {
          int o = o0 + i + hh * 8;
          int n = nB + tn * 16 + r;
          gF[fBase + (size_t)(o - 2 * CI_) * N_ + n] = (half_t)(acc[to][tn][i] + biasCat[o]);
        }
    }
  }
}

// ---------------------------------------------------- 2x2x2 maxpool on phi / g
__global__ void pool_kernel(const half_t* __restrict__ phiF, const half_t* __restrict__ gF,
                            half_t* __restrict__ phi_p, half_t* __restrict__ g_pT) {
  int idx = blockIdx.x * 256 + threadIdx.x;            // exact: B*CI*M threads
  int m = idx % M_;
  int ci = (idx / M_) % CI_;
  int b = idx / (M_ * CI_);
  int tp = m >> 8, hp = (m >> 4) & 15, wp = m & 15;
  size_t base = ((size_t)b * CI_ + ci) * N_ + (size_t)(2 * tp) * (H_ * W_) + (2 * hp) * W_ + 2 * wp;
  float mp = -1e30f, mg = -1e30f;
#pragma unroll
  for (int dt = 0; dt < 2; ++dt)
#pragma unroll
    for (int dh = 0; dh < 2; ++dh)
#pragma unroll
      for (int dw = 0; dw < 2; ++dw) {
        size_t off = base + (size_t)dt * (H_ * W_) + dh * W_ + dw;
        mp = fmaxf(mp, (float)phiF[off]);
        mg = fmaxf(mg, (float)gF[off]);
      }
  phi_p[((size_t)b * M_ + m) * CI_ + ci] = (half_t)mp;   // (b, M, CI): K-frag layout
  g_pT[((size_t)b * CI_ + ci) * M_ + m] = (half_t)mg;    // (b, CI, M): V-frag layout
}

// ------------------------- flash attention: Y = softmax(Q Kt) V, per 16-row strip
__global__ __launch_bounds__(256, 1) void attn_kernel(
    const half_t* __restrict__ thetaT, const half_t* __restrict__ phi_p,
    const half_t* __restrict__ g_pT, half_t* __restrict__ yws) {
  __shared__ half_t pshare[8][16 * 32];
#if HAVE_TDM
  __shared__ __align__(32) half_t vshare[2 * CI_ * 32];  // 2 x 16KB V tiles (double buffer)
#endif
  int lane = threadIdx.x & 31, wave = threadIdx.x >> 5;
  int r = lane & 15, hh = lane >> 4;
  int b = blockIdx.y;
  int nBase = blockIdx.x * 128 + wave * 16;
  const half_t* Q  = thetaT + (size_t)b * N_ * CI_;  // (N, CI)
  const half_t* Kp = phi_p  + (size_t)b * M_ * CI_;  // (M, CI)
  const half_t* Vt = g_pT   + (size_t)b * CI_ * M_;  // (CI, M)
  half_t* pwave = &pshare[wave][0];

#if HAVE_TDM
  // Pre-issue V tiles 0 and 1: the tensor DMA runs underneath all of pass 1.
  unsigned int vbase = (unsigned int)(uintptr_t)vshare;
  if (wave == 0) {
    tdm_load_2d(vbase,                    Vt,      32, CI_, M_);
    tdm_load_2d(vbase + CI_ * 32 * 2,     Vt + 32, 32, CI_, M_);
  }
#endif

  // Q fragments for this wave's 16-row strip: kept in VGPRs for both passes.
  v16h qf[CI_ / 32];
#pragma unroll
  for (int kk = 0; kk < CI_ / 32; ++kk)
    qf[kk] = load_a_frag(Q, CI_, nBase + r, kk * 32, hh);

  // ---- pass 1: online row max & sum of exp over all 64 m-tiles
  float rmax[8], rsum[8];
#pragma unroll
  for (int i = 0; i < 8; ++i) { rmax[i] = -1e30f; rsum[i] = 0.f; }

  for (int mt = 0; mt < M_ / 16; ++mt) {
    v8f s = {};
#pragma unroll
    for (int kk = 0; kk < CI_ / 32; ++kk) {
      v16h bb = load_b_frag(Kp, CI_, mt * 16 + r, kk * 32, hh);
      s = wmma_f32(qf[kk], bb, s);
    }
#pragma unroll
    for (int i = 0; i < 8; ++i) {
      float tm = s[i];
      tm = fmaxf(tm, __shfl_xor(tm, 8));
      tm = fmaxf(tm, __shfl_xor(tm, 4));
      tm = fmaxf(tm, __shfl_xor(tm, 2));
      tm = fmaxf(tm, __shfl_xor(tm, 1));
      float nm = fmaxf(rmax[i], tm);
      float ts = __expf(s[i] - nm);
      ts += __shfl_xor(ts, 8);
      ts += __shfl_xor(ts, 4);
      ts += __shfl_xor(ts, 2);
      ts += __shfl_xor(ts, 1);
      rsum[i] = rsum[i] * __expf(rmax[i] - nm) + ts;
      rmax[i] = nm;
    }
  }

  // ---- pass 2: recompute S, P = exp(S-max), accumulate Y += P V
  v8f yacc[16] = {};
  for (int mc = 0; mc < NTV_; ++mc) {
#if HAVE_TDM
    // Issue tile mc+1 into buffer (mc+1)&1 (last read in iteration mc-1,
    // which ended with a barrier) -> DMA overlaps this iteration's S compute.
    if (wave == 0 && mc >= 1 && mc + 1 < NTV_)
      tdm_load_2d(vbase + (unsigned int)(((mc + 1) & 1) * CI_ * 32 * 2),
                  Vt + (mc + 1) * 32, 32, CI_, M_);
#endif
#pragma unroll
    for (int j = 0; j < 2; ++j) {
      v8f s = {};
#pragma unroll
      for (int kk = 0; kk < CI_ / 32; ++kk) {
        v16h bb = load_b_frag(Kp, CI_, mc * 32 + j * 16 + r, kk * 32, hh);
        s = wmma_f32(qf[kk], bb, s);
      }
#pragma unroll
      for (int i = 0; i < 8; ++i)
        pwave[(i + hh * 8) * 32 + j * 16 + r] = (half_t)__expf(s[i] - rmax[i]);
    }
#if HAVE_TDM
    // TENSORcnt completes in order: <=1 outstanding => tile mc is resident.
    if (mc + 1 < NTV_) __builtin_amdgcn_s_wait_tensorcnt(1);
    else               __builtin_amdgcn_s_wait_tensorcnt(0);
#endif
    __syncthreads();  // P tile (cross-lane) + V tile visible to all waves
    AFrag pa;
    pa.h[0] = *(const v8h*)&pwave[r * 32 + hh * 16];
    pa.h[1] = *(const v8h*)&pwave[r * 32 + hh * 16 + 8];
#if HAVE_TDM
    const half_t* vtile = &vshare[(mc & 1) * CI_ * 32];
#endif
#pragma unroll
    for (int ct = 0; ct < 16; ++ct) {
#if HAVE_TDM
      v16h bv = *(const v16h*)&vtile[(ct * 16 + r) * 32 + hh * 16];
#else
      v16h bv = *(const v16h*)(Vt + (size_t)(ct * 16 + r) * M_ + mc * 32 + hh * 16);
#endif
      yacc[ct] = wmma_f32(pa.v, bv, yacc[ct]);
    }
    __syncthreads();  // all reads done before the next TDM overwrites this buffer
  }

  // ---- epilogue: divide by softmax denominator, store Y (b, N, CI) f16
  size_t yBase = (size_t)b * N_ * CI_;
#pragma unroll
  for (int ct = 0; ct < 16; ++ct) {
#pragma unroll
    for (int i = 0; i < 8; ++i) {
      int n = nBase + i + hh * 8;
      int ci = ct * 16 + r;
      yws[yBase + (size_t)n * CI_ + ci] = (half_t)(yacc[ct][i] / rsum[i]);
    }
  }
}

// --------------------------- GEMM Wz: [512,256] x [256,N] -> z (b,C,N) fp32
__global__ __launch_bounds__(256, 1) void gemm_wz_kernel(
    const half_t* __restrict__ wzh, const float* __restrict__ wzb,
    const half_t* __restrict__ yws, float* __restrict__ z) {
  int lane = threadIdx.x & 31, wave = threadIdx.x >> 5;
  int r = lane & 15, hh = lane >> 4;
  int oB = blockIdx.x * 64;                 // scalar
  int nB = blockIdx.y * 512 + wave * 64;    // waves partition n
  int b = blockIdx.z;
  const half_t* Y = yws + (size_t)b * N_ * CI_;   // (N, CI)

  v8f acc[4][4] = {};
  for (int kk = 0; kk < CI_ / 32; ++kk) {
    int k0 = kk * 32;
    v16h a[4], bb[4];
#pragma unroll
    for (int to = 0; to < 4; ++to)
      a[to] = load_a_frag(wzh, CI_, oB + to * 16 + r, k0, hh);
#pragma unroll
    for (int tn = 0; tn < 4; ++tn)
      bb[tn] = load_b_frag(Y, CI_, nB + tn * 16 + r, k0, hh);
#pragma unroll
    for (int to = 0; to < 4; ++to)
#pragma unroll
      for (int tn = 0; tn < 4; ++tn)
        acc[to][tn] = wmma_f32(a[to], bb[tn], acc[to][tn]);
  }
  size_t zBase = (size_t)b * C_ * N_;
#pragma unroll
  for (int to = 0; to < 4; ++to)
#pragma unroll
    for (int tn = 0; tn < 4; ++tn)
#pragma unroll
      for (int i = 0; i < 8; ++i) {
        int o = oB + to * 16 + i + hh * 8;
        int n = nB + tn * 16 + r;
        z[zBase + (size_t)o * N_ + n] = acc[to][tn][i] + wzb[o];
      }
}

// ------------------------------------------------ BN stats: mean & rstd per channel
__global__ __launch_bounds__(1024) void bn_stats_kernel(const float* __restrict__ z,
                                                        float* __restrict__ meanArr,
                                                        float* __restrict__ rstdArr) {
  int c = blockIdx.x;
  __shared__ float s1[1024], s2[1024];
  float a1 = 0.f, a2 = 0.f;
  for (int t = threadIdx.x; t < B_ * N_; t += 1024) {
    int b = t >> 13;
    int n = t & (N_ - 1);
    float v = z[((size_t)b * C_ + c) * N_ + n];
    a1 += v; a2 += v * v;
  }
  s1[threadIdx.x] = a1; s2[threadIdx.x] = a2;
  __syncthreads();
  for (int s = 512; s > 0; s >>= 1) {
    if ((int)threadIdx.x < s) {
      s1[threadIdx.x] += s1[threadIdx.x + s];
      s2[threadIdx.x] += s2[threadIdx.x + s];
    }
    __syncthreads();
  }
  if (threadIdx.x == 0) {
    const float inv = 1.0f / (float)(B_ * N_);
    float mean = s1[0] * inv;
    float var = s2[0] * inv - mean * mean;
    meanArr[c] = mean;
    rstdArr[c] = rsqrtf(var + 1e-5f);
  }
}

// ---------------------------------------- normalize + affine + residual -> out
__global__ void finalize_kernel(const float* __restrict__ z, const float* __restrict__ x,
                                const float* __restrict__ meanArr, const float* __restrict__ rstdArr,
                                const float* __restrict__ gamma, const float* __restrict__ beta,
                                float* __restrict__ out) {
  size_t idx = (size_t)blockIdx.x * 256 + threadIdx.x;
  if (idx >= (size_t)B_ * C_ * N_) return;
  int c = (int)((idx / N_) % C_);
  out[idx] = (z[idx] - meanArr[c]) * rstdArr[c] * gamma[c] + beta[c] + x[idx];
}

extern "C" void kernel_launch(void* const* d_in, const int* in_sizes, int n_in,
                              void* d_out, int out_size, void* d_ws, size_t ws_size,
                              hipStream_t stream) {
  const float* x   = (const float*)d_in[0];
  const float* tw  = (const float*)d_in[1];
  const float* tb  = (const float*)d_in[2];
  const float* pw  = (const float*)d_in[3];
  const float* pb  = (const float*)d_in[4];
  const float* gw  = (const float*)d_in[5];
  const float* gb  = (const float*)d_in[6];
  const float* wzw = (const float*)d_in[7];
  const float* wzb = (const float*)d_in[8];
  const float* bn_gamma = (const float*)d_in[9];
  const float* bn_beta  = (const float*)d_in[10];
  float* out = (float*)d_out;

  char* p = (char*)d_ws;
  auto alloc = [&](size_t bytes) -> char* {
    char* ret = p;
    p += (bytes + 255) & ~(size_t)255;
    return ret;
  };
  half_t* xh      = (half_t*)alloc((size_t)B_ * N_ * C_ * sizeof(half_t));
  half_t* Wcat    = (half_t*)alloc((size_t)O3_ * C_ * sizeof(half_t));
  float*  biasCat = (float*) alloc((size_t)O3_ * sizeof(float));
  half_t* wzh     = (half_t*)alloc((size_t)C_ * CI_ * sizeof(half_t));
  half_t* thetaT  = (half_t*)alloc((size_t)B_ * N_ * CI_ * sizeof(half_t));
  half_t* phiF    = (half_t*)alloc((size_t)B_ * CI_ * N_ * sizeof(half_t));
  half_t* gF      = (half_t*)alloc((size_t)B_ * CI_ * N_ * sizeof(half_t));
  half_t* phi_p   = (half_t*)alloc((size_t)B_ * M_ * CI_ * sizeof(half_t));
  half_t* g_pT    = (half_t*)alloc((size_t)B_ * CI_ * M_ * sizeof(half_t));
  half_t* yws     = (half_t*)alloc((size_t)B_ * N_ * CI_ * sizeof(half_t));
  float*  zws     = (float*) alloc((size_t)B_ * C_ * N_ * sizeof(float));
  float*  meanArr = (float*) alloc((size_t)C_ * sizeof(float));
  float*  rstdArr = (float*) alloc((size_t)C_ * sizeof(float));

  // 1. weights -> f16 (concatenated) + bias concat
  prep_kernel<<<(O3_ * C_ + 255) / 256, 256, 0, stream>>>(tw, tb, pw, pb, gw, gb, wzw,
                                                          Wcat, biasCat, wzh);
  // 2. x (b,C,N) fp32 -> xh (b,N,C) f16
  transpose_kernel<<<dim3(N_ / 32, C_ / 32, B_), dim3(32, 8), 0, stream>>>(x, xh);
  // 3. fused theta/phi/g projection GEMM (64x64 wave tiles)
  gemm_proj_kernel<<<dim3(O3_ / 64, N_ / 512, B_), 256, 0, stream>>>(Wcat, biasCat, xh,
                                                                     thetaT, phiF, gF);
  // 4. 2x2x2 maxpool on phi / g
  pool_kernel<<<(B_ * CI_ * M_) / 256, 256, 0, stream>>>(phiF, gF, phi_p, g_pT);
  // 5. flash attention (double-buffered TDM V tiles when available)
  attn_kernel<<<dim3(N_ / 128, B_), 256, 0, stream>>>(thetaT, phi_p, g_pT, yws);
  // 6. output projection Wz (64x64 wave tiles)
  gemm_wz_kernel<<<dim3(C_ / 64, N_ / 512, B_), 256, 0, stream>>>(wzh, wzb, yws, zws);
  // 7. BatchNorm statistics
  bn_stats_kernel<<<C_, 1024, 0, stream>>>(zws, meanArr, rstdArr);
  // 8. normalize + affine + residual
  finalize_kernel<<<(int)(((size_t)B_ * C_ * N_ + 255) / 256), 256, 0, stream>>>(
      zws, x, meanArr, rstdArr, bn_gamma, bn_beta, out);
}
